// CLIPAttention_74071005987594
// MI455X (gfx1250) — matrix operand order repaired and checked by reading, not compiled, not run on hardware
//
#include <hip/hip_runtime.h>
#include <cstdint>
#include <cstddef>

#define EMBED   1024
#define NHEADS  16
#define HDIM    64
#define BATCH   32
#define SEQ     577
#define SPAD    608                 // SEQ padded to multiple of 32
#define MTOT    (BATCH * SEQ)       // 18464
#define QK_SCALE 0.125f             // 64^-0.5

typedef __bf16 bf16_t;
typedef __bf16 v16bf __attribute__((ext_vector_type(16)));
typedef __bf16 v8bf  __attribute__((ext_vector_type(8)));
typedef float  v8f   __attribute__((ext_vector_type(8)));
typedef unsigned int u32x4 __attribute__((ext_vector_type(4)));
typedef int i32x4 __attribute__((ext_vector_type(4)));
typedef int i32x8 __attribute__((ext_vector_type(8)));

// ---------------------------------------------------------------------------
// gfx1250 async / TDM feature detection (guarded; register-path fallbacks)
// ---------------------------------------------------------------------------
#if __has_builtin(__builtin_amdgcn_global_load_async_to_lds_b128)
#define HAS_ASYNC 1
#else
#define HAS_ASYNC 0
#endif
#if __has_builtin(__builtin_amdgcn_tensor_load_to_lds)
#define HAS_TDM 1
#else
#define HAS_TDM 0
#endif

// async-to-LDS builtin wants int4* operands: AS1 (global) src, AS3 (LDS) dst
typedef __attribute__((address_space(1))) i32x4* gl_i4p;
typedef __attribute__((address_space(3))) i32x4* lds_i4p;

// copy 8 bf16 (16B) global -> LDS
__device__ __forceinline__ void cp8_async(bf16_t* dst_lds, const bf16_t* src_glob) {
#if HAS_ASYNC
  // flat-LDS aperture keeps the LDS byte offset in addr[31:0]
  __builtin_amdgcn_global_load_async_to_lds_b128(
      (gl_i4p)(uintptr_t)src_glob,
      (lds_i4p)(uint32_t)(uintptr_t)dst_lds, 0, 0);
#else
  *(v8bf*)dst_lds = *(const v8bf*)src_glob;
#endif
}

template <int N>
__device__ __forceinline__ void wait_async() {
#if HAS_ASYNC
#if __has_builtin(__builtin_amdgcn_s_wait_asynccnt)
  __builtin_amdgcn_s_wait_asynccnt((short)N);
#else
  asm volatile("s_wait_asynccnt %0" ::"i"(N) : "memory");
#endif
#endif
}

template <int N>
__device__ __forceinline__ void wait_tensor() {
#if HAS_TDM
#if __has_builtin(__builtin_amdgcn_s_wait_tensorcnt)
  __builtin_amdgcn_s_wait_tensorcnt((short)N);
#else
  asm volatile("s_wait_tensorcnt %0" ::"i"(N) : "memory");
#endif
#endif
}

#if HAS_TDM
// TDM 2-D tile load: bf16 tile [tile_rows x tile_cols] from a row-major
// tensor with stride row_stride (elements) into a linear LDS region.
// D# per CDNA5 ISA ch8: group0 {count,lds,global,type=2}, group1 {dims}.
__device__ __forceinline__ void tdm_load_2d(uint32_t lds_off, const bf16_t* gsrc,
                                            int tile_cols, int tile_rows,
                                            int tensor_w, int tensor_h,
                                            int row_stride) {
  const uint64_t ga = (uint64_t)(uintptr_t)gsrc;
  u32x4 g0;
  g0[0] = 1u;                                    // count=1, user descriptor
  g0[1] = lds_off;                               // lds_addr (bytes)
  g0[2] = (uint32_t)ga;                          // global_addr[31:0]
  g0[3] = (uint32_t)(ga >> 32) | 0x80000000u;    // global_addr[56:32] | type=2
  i32x8 g1;
  g1[0] = 0x10000;                               // data_size=1 (2 bytes)
  g1[1] = (tensor_w & 0xFFFF) << 16;             // tensor_dim0[15:0] @bit48
  g1[2] = ((tensor_w >> 16) & 0xFFFF) | ((tensor_h & 0xFFFF) << 16);
  g1[3] = ((tensor_h >> 16) & 0xFFFF) | (tile_cols << 16);   // tile_dim0
  g1[4] = tile_rows & 0xFFFF;                    // tile_dim1 (tile_dim2=0)
  g1[5] = row_stride;                            // tensor_dim0_stride[31:0]
  g1[6] = 0;
  g1[7] = 0;
  const i32x4 z4 = {0, 0, 0, 0};
#if __clang_major__ >= 23
  const i32x8 z8 = {0, 0, 0, 0, 0, 0, 0, 0};
  __builtin_amdgcn_tensor_load_to_lds(g0, g1, z4, z4, z8, 0);
#else
  __builtin_amdgcn_tensor_load_to_lds(g0, g1, z4, z4, 0);
#endif
}
#endif

// ---------------------------------------------------------------------------
// misc helpers
// ---------------------------------------------------------------------------
__device__ __forceinline__ v16bf pack16(const bf16_t* lo, const bf16_t* hi) {
  v8bf a = *(const v8bf*)lo;
  v8bf b = *(const v8bf*)hi;
  v16bf r;
#pragma unroll
  for (int i = 0; i < 8; ++i) { r[i] = a[i]; r[i + 8] = b[i]; }
  return r;
}

template <int P>
__device__ __forceinline__ float lane_xor(float v) {
#if __has_builtin(__builtin_amdgcn_ds_swizzle)
  return __int_as_float(__builtin_amdgcn_ds_swizzle(__float_as_int(v), 0x1F | (P << 10)));
#else
  return __shfl_xor(v, P, 32);
#endif
}
__device__ __forceinline__ float redmax16(float x) {
  x = fmaxf(x, lane_xor<1>(x));
  x = fmaxf(x, lane_xor<2>(x));
  x = fmaxf(x, lane_xor<4>(x));
  x = fmaxf(x, lane_xor<8>(x));
  return x;
}
__device__ __forceinline__ float redsum16(float x) {
  x += lane_xor<1>(x);
  x += lane_xor<2>(x);
  x += lane_xor<4>(x);
  x += lane_xor<8>(x);
  return x;
}

// ---------------------------------------------------------------------------
// fp32 -> bf16
// ---------------------------------------------------------------------------
__global__ __launch_bounds__(256) void cvt_f32_bf16(const float* __restrict__ in,
                                                    bf16_t* __restrict__ out, long n) {
  long i = (long)blockIdx.x * blockDim.x + threadIdx.x;
  const long stride = (long)gridDim.x * blockDim.x;
  for (; i < n; i += stride) out[i] = (bf16_t)in[i];
}

// ---------------------------------------------------------------------------
// Tiled WMMA GEMM, double-buffered async staging:
//   Y[m,n] = sum_k A[m,k] * W[n,k] + bias[n]
// Block 256 threads (8 waves); BM=256 (32 rows/wave), BN=64, BK=32;
// per wave per k-step: 8 WMMAs.
// MODE 0: bf16 out in [B,H,SPAD,HDIM] layout, scaled; MODE 1: fp32 [MTOT,EMBED]
// ---------------------------------------------------------------------------
#if HAS_TDM
#define GEMM_APEND 4   // 4 async A ops/thread outstanding
#else
#define GEMM_APEND 5   // + 1 W op/thread
#endif

template <int MODE>
__global__ __launch_bounds__(256) void gemm_bias_wmma(
    const bf16_t* __restrict__ A, const bf16_t* __restrict__ W,
    const float* __restrict__ bias, void* __restrict__ Out, float scale) {
  __shared__ __align__(16) bf16_t As[2][256][32];
  __shared__ __align__(16) bf16_t Ws[2][64][32];
  const int t = threadIdx.x;
  const int wave = t >> 5, lane = t & 31;
  const int m0 = blockIdx.x * 256;
  const int n0 = blockIdx.y * 64;

  // A staging: thread t owns row t of the 256x32 tile (4 x b128)
  const int agrow = m0 + t;
  const bf16_t* aptr = A + (size_t)agrow * EMBED;
  // W staging fallback: row = t>>2, col = (t&3)*8
  const int wrow = t >> 2, wcol = (t & 3) * 8;
  const bf16_t* wptr = W + (size_t)(n0 + wrow) * EMBED + wcol;

  auto issue = [&](int kb, int buf) {
    if (agrow < MTOT) {
#pragma unroll
      for (int c = 0; c < 32; c += 8) cp8_async(&As[buf][t][c], aptr + kb + c);
    }
#if HAS_TDM
    if (wave == 0)
      tdm_load_2d((uint32_t)(uintptr_t)&Ws[buf][0][0],
                  W + (size_t)n0 * EMBED + kb, /*tile_cols=*/32, /*tile_rows=*/64,
                  /*tensor_w=*/EMBED, /*tensor_h=*/EMBED, /*row_stride=*/EMBED);
#else
    cp8_async(&Ws[buf][wrow][wcol], wptr + kb);
#endif
  };

  v8f acc[2][4] = {};
  issue(0, 0);
  const int NK = EMBED / 32;
  for (int i = 0; i < NK; ++i) {
    const bool has_next = (i + 1) < NK;
    if (has_next) {
      issue((i + 1) * 32, (i + 1) & 1);
      wait_async<GEMM_APEND>();          // previous batch complete
#if HAS_TDM
      if (wave == 0) wait_tensor<1>();
#endif
    } else {
      wait_async<0>();
#if HAS_TDM
      if (wave == 0) wait_tensor<0>();
#endif
    }
    __syncthreads();

    const int buf = i & 1;
    const int mrow = wave * 32 + (lane & 15);
    const int koff = (lane >> 4) * 8;
    const int kr = (lane >> 4) * 16;
    const v16bf a0 = pack16(&As[buf][mrow][koff], &As[buf][mrow][16 + koff]);
    const v16bf a1 = pack16(&As[buf][mrow + 16][koff], &As[buf][mrow + 16][16 + koff]);
#pragma unroll
    for (int ns = 0; ns < 4; ++ns) {
      const int wr = ns * 16 + (lane & 15);
      const v16bf b = pack16(&Ws[buf][wr][koff], &Ws[buf][wr][16 + koff]);
      acc[0][ns] = __builtin_amdgcn_wmma_f32_16x16x32_bf16(
          false, a0, false, b, (short)0, acc[0][ns], false, false);
      acc[1][ns] = __builtin_amdgcn_wmma_f32_16x16x32_bf16(
          false, a1, false, b, (short)0, acc[1][ns], false, false);
    }
    __syncthreads();
  }

  // epilogue (C layout: vgpr r -> row r + 8*(lane>>4), col = lane&15)
  const int rb = (lane >> 4) * 8;
  const int ncol = lane & 15;
#pragma unroll
  for (int mt = 0; mt < 2; ++mt) {
#pragma unroll
    for (int ns = 0; ns < 4; ++ns) {
      const int n = n0 + ns * 16 + ncol;
      const float bv = bias[n];
#pragma unroll
      for (int r = 0; r < 8; ++r) {
        const int m = m0 + wave * 32 + mt * 16 + r + rb;
        if (m < MTOT) {
          const float v = (acc[mt][ns][r] + bv) * scale;
          if (MODE == 0) {
            const int b_ = m / SEQ, s_ = m % SEQ;
            const int h_ = n >> 6, dh = n & 63;
            ((bf16_t*)Out)[(((size_t)b_ * NHEADS + h_) * SPAD + s_) * HDIM + dh] = (bf16_t)v;
          } else {
            ((float*)Out)[(size_t)m * EMBED + n] = v;
          }
        }
      }
    }
  }
}

// ---------------------------------------------------------------------------
// Flash attention: one wave per 16-query tile, 4 waves/block share K/V LDS.
// Q pre-scaled. Q/K/V layout [B*H, SPAD, HDIM] bf16. Output bf16 [B,S,EMBED].
// ---------------------------------------------------------------------------
__global__ __launch_bounds__(128) void flash_attn_wmma(
    const bf16_t* __restrict__ Q, const bf16_t* __restrict__ K,
    const bf16_t* __restrict__ V, bf16_t* __restrict__ AO) {
  __shared__ __align__(16) bf16_t Ks[32][HDIM];   // [key][dh]
  __shared__ __align__(16) bf16_t Vt[HDIM][32];   // [dh][key]
  __shared__ __align__(16) bf16_t Ps[4][16][32];  // per-wave P staging

  const int t = threadIdx.x;
  const int wave = t >> 5, lane = t & 31;
  const int bh = blockIdx.y;
  const int b_ = bh >> 4, h_ = bh & 15;
  const int qt = blockIdx.x * 4 + wave;
  const bool active = (qt * 16 < SEQ);
  const int q0 = qt * 16;

  const bf16_t* Qb = Q + (size_t)bh * SPAD * HDIM;
  const bf16_t* Kb = K + (size_t)bh * SPAD * HDIM;
  const bf16_t* Vb = V + (size_t)bh * SPAD * HDIM;

  v16bf aq0, aq1;
#pragma unroll
  for (int i = 0; i < 16; ++i) { aq0[i] = (bf16_t)0.0f; aq1[i] = (bf16_t)0.0f; }
  if (active) {
    const int row = q0 + (lane & 15);
    const int koff = (lane >> 4) * 8;
    const bf16_t* qp = Qb + (size_t)row * HDIM;
    aq0 = pack16(qp + koff, qp + 16 + koff);
    aq1 = pack16(qp + 32 + koff, qp + 48 + koff);
  }

  float mrow[8], lrow[8];
#pragma unroll
  for (int r = 0; r < 8; ++r) { mrow[r] = -1.0e30f; lrow[r] = 0.0f; }
  v8f oacc[4] = {};

  for (int kc = 0; kc < SPAD; kc += 32) {
    // stage K (async path) and V-transposed (register path)
    {
      const int kk = t >> 2;
      const int d0 = (t & 3) * 16;
      const bf16_t* kp = Kb + (size_t)(kc + kk) * HDIM + d0;
      cp8_async(&Ks[kk][d0], kp);
      cp8_async(&Ks[kk][d0 + 8], kp + 8);
      const bf16_t* vp = Vb + (size_t)(kc + kk) * HDIM + d0;
      const v8bf v0 = *(const v8bf*)vp;
      const v8bf v1 = *(const v8bf*)(vp + 8);
#pragma unroll
      for (int i = 0; i < 8; ++i) {
        Vt[d0 + i][kk] = v0[i];
        Vt[d0 + 8 + i][kk] = v1[i];
      }
    }
    wait_async<0>();
    __syncthreads();

    if (active) {
      v8f c[2];
      const int krg = (lane >> 4) * 16;
#pragma unroll
      for (int tile = 0; tile < 2; ++tile) {
        const int keyc = tile * 16 + (lane & 15);
        const v16bf bk0 = pack16(&Ks[keyc][krg / 2], &Ks[keyc][16 + krg / 2]);
        const v16bf bk1 = pack16(&Ks[keyc][32 + krg / 2], &Ks[keyc][48 + krg / 2]);
        v8f cc = {};
        cc = __builtin_amdgcn_wmma_f32_16x16x32_bf16(false, aq0, false, bk0, (short)0, cc, false, false);
        cc = __builtin_amdgcn_wmma_f32_16x16x32_bf16(false, aq1, false, bk1, (short)0, cc, false, false);
        if (kc + keyc >= SEQ) {
#pragma unroll
          for (int r = 0; r < 8; ++r) cc[r] = -1.0e30f;
        }
        c[tile] = cc;
      }

      float alpha[8];
#pragma unroll
      for (int r = 0; r < 8; ++r) {
        float mc = redmax16(fmaxf(c[0][r], c[1][r]));
        const float mn = fmaxf(mrow[r], mc);
        alpha[r] = __expf(mrow[r] - mn);
        mrow[r] = mn;
        const float p0 = __expf(c[0][r] - mn);
        const float p1 = __expf(c[1][r] - mn);
        c[0][r] = p0;
        c[1][r] = p1;
        lrow[r] = lrow[r] * alpha[r] + redsum16(p0 + p1);
      }
#pragma unroll
      for (int ds = 0; ds < 4; ++ds)
#pragma unroll
        for (int r = 0; r < 8; ++r) oacc[ds][r] *= alpha[r];

      // C-layout -> A-layout for P via per-wave LDS staging
      {
        const int rb = (lane >> 4) * 8;
        const int col = lane & 15;
#pragma unroll
        for (int tile = 0; tile < 2; ++tile)
#pragma unroll
          for (int r = 0; r < 8; ++r)
            Ps[wave][r + rb][tile * 16 + col] = (bf16_t)c[tile][r];
      }
      asm volatile("s_wait_dscnt 0x0" ::: "memory");
      const int prow = lane & 15;
      const int pk = (lane >> 4) * 8;
      const v16bf ap = pack16(&Ps[wave][prow][pk], &Ps[wave][prow][16 + pk]);

#pragma unroll
      for (int ds = 0; ds < 4; ++ds) {
        const int dh = ds * 16 + (lane & 15);
        const v16bf bv = pack16(&Vt[dh][pk], &Vt[dh][16 + pk]);
        oacc[ds] = __builtin_amdgcn_wmma_f32_16x16x32_bf16(
            false, ap, false, bv, (short)0, oacc[ds], false, false);
      }
    }
    __syncthreads();
  }

  if (active) {
    const int rb = (lane >> 4) * 8;
    const int col = lane & 15;
#pragma unroll
    for (int ds = 0; ds < 4; ++ds) {
      const int dh = ds * 16 + col;
#pragma unroll
      for (int r = 0; r < 8; ++r) {
        const int q = q0 + r + rb;
        if (q < SEQ)
          AO[((size_t)b_ * SEQ + q) * EMBED + h_ * HDIM + dh] =
              (bf16_t)(oacc[ds][r] / lrow[r]);
      }
    }
  }
}

// ---------------------------------------------------------------------------
// host orchestration
// ---------------------------------------------------------------------------
extern "C" void kernel_launch(void* const* d_in, const int* in_sizes, int n_in,
                              void* d_out, int out_size, void* d_ws, size_t ws_size,
                              hipStream_t stream) {
  (void)in_sizes; (void)n_in; (void)out_size; (void)ws_size;
  const float* hs  = (const float*)d_in[0];
  const float* q_w = (const float*)d_in[1];
  const float* q_b = (const float*)d_in[2];
  const float* k_w = (const float*)d_in[3];
  const float* k_b = (const float*)d_in[4];
  const float* v_w = (const float*)d_in[5];
  const float* v_b = (const float*)d_in[6];
  const float* o_w = (const float*)d_in[7];
  const float* o_b = (const float*)d_in[8];

  char* p = (char*)d_ws;
  bf16_t* hsb = (bf16_t*)p; p += (size_t)MTOT * EMBED * sizeof(bf16_t);   // reused as AO
  bf16_t* qwb = (bf16_t*)p; p += (size_t)EMBED * EMBED * sizeof(bf16_t);
  bf16_t* kwb = (bf16_t*)p; p += (size_t)EMBED * EMBED * sizeof(bf16_t);
  bf16_t* vwb = (bf16_t*)p; p += (size_t)EMBED * EMBED * sizeof(bf16_t);
  bf16_t* owb = (bf16_t*)p; p += (size_t)EMBED * EMBED * sizeof(bf16_t);
  const size_t qkv_elems = (size_t)BATCH * NHEADS * SPAD * HDIM;
  bf16_t* Qb = (bf16_t*)p; p += qkv_elems * sizeof(bf16_t);
  bf16_t* Kb = (bf16_t*)p; p += qkv_elems * sizeof(bf16_t);
  bf16_t* Vb = (bf16_t*)p; p += qkv_elems * sizeof(bf16_t);

  cvt_f32_bf16<<<4096, 256, 0, stream>>>(hs, hsb, (long)MTOT * EMBED);
  cvt_f32_bf16<<<1024, 256, 0, stream>>>(q_w, qwb, (long)EMBED * EMBED);
  cvt_f32_bf16<<<1024, 256, 0, stream>>>(k_w, kwb, (long)EMBED * EMBED);
  cvt_f32_bf16<<<1024, 256, 0, stream>>>(v_w, vwb, (long)EMBED * EMBED);
  cvt_f32_bf16<<<1024, 256, 0, stream>>>(o_w, owb, (long)EMBED * EMBED);

  (void)hipMemsetAsync(Qb, 0, 3 * qkv_elems * sizeof(bf16_t), stream);

  const dim3 ggrid((MTOT + 255) / 256, EMBED / 64, 1);
  gemm_bias_wmma<0><<<ggrid, 256, 0, stream>>>(hsb, qwb, q_b, (void*)Qb, QK_SCALE);
  gemm_bias_wmma<0><<<ggrid, 256, 0, stream>>>(hsb, kwb, k_b, (void*)Kb, 1.0f);
  gemm_bias_wmma<0><<<ggrid, 256, 0, stream>>>(hsb, vwb, v_b, (void*)Vb, 1.0f);

  const int qtiles = (SEQ + 15) / 16;                 // 37
  const dim3 agrid((qtiles + 3) / 4, BATCH * NHEADS); // 10 x 512
  flash_attn_wmma<<<agrid, 128, 0, stream>>>(Qb, Kb, Vb, hsb);

  gemm_bias_wmma<1><<<ggrid, 256, 0, stream>>>(hsb, owb, o_b, d_out, 1.0f);
}
